// DQNxLSTM_52733608460768
// MI455X (gfx1250) — matrix operand order repaired and checked by reading, not compiled, not run on hardware
//
#include <hip/hip_runtime.h>

typedef __attribute__((ext_vector_type(16))) __bf16 v16bf;
typedef __attribute__((ext_vector_type(8)))  float  v8f;
typedef __attribute__((ext_vector_type(4)))  unsigned int v4u;

constexpr int BATCH = 32;
constexpr int SEQ   = 2048;
constexpr int CIN   = 256;
constexpr int HID   = 512;
constexpr int FH    = 4 * HID;       // 2048 gate columns
constexpr int HB    = 32;            // hidden units per workgroup
constexpr int NWG   = HID / HB;      // 16 workgroups per layer
constexpr int BH    = BATCH * HID;   // 16384 elements of one h state

__device__ __forceinline__ unsigned short f2bf(float x) {
  unsigned u = __float_as_uint(x);
  u += 0x7FFFu + ((u >> 16) & 1u);   // round-to-nearest-even
  return (unsigned short)(u >> 16);
}
__device__ __forceinline__ unsigned pack2(float lo, float hi) {
  return (unsigned)f2bf(lo) | ((unsigned)f2bf(hi) << 16);
}
__device__ __forceinline__ float fast_tanh(float x) {
  x = fminf(fmaxf(x, -15.f), 15.f);
  float e = __expf(2.f * x);
  return (e - 1.f) * __builtin_amdgcn_rcpf(e + 1.f);
}
__device__ __forceinline__ float fast_sig(float x) {
  return __builtin_amdgcn_rcpf(1.f + __expf(-x));
}

union AFrag { v16bf v; v4u q[2]; };
union BFrag { v16bf v; v4u q[2]; };

__global__ void init_ws(unsigned short* hbuf, unsigned int* cnt) {
  const int stride = gridDim.x * blockDim.x;
  int i = blockIdx.x * blockDim.x + threadIdx.x;
  for (int k = i; k < 4 * BH; k += stride) hbuf[k] = 0;   // 2 layers x 2 h buffers
  for (int k = i; k < 2 * SEQ; k += stride) cnt[k] = 0;   // per-step barrier counters
}

// One-time: x [B,SEQ,CIN] fp32 -> xbf [SEQ,B,CIN] bf16 (transpose + convert),
// so both layers feed the scan with the identical [t][batch][K] bf16 layout.
__global__ void prep_x(const float* __restrict__ x, unsigned short* __restrict__ xbf) {
  const size_t total = (size_t)BATCH * SEQ * (CIN / 8);
  size_t i = (size_t)blockIdx.x * blockDim.x + threadIdx.x;
  if (i >= total) return;
  const size_t c8 = (i % (CIN / 8)) * 8;
  const size_t rt = i / (CIN / 8);
  const size_t t  = rt % SEQ;
  const size_t r  = rt / SEQ;
  const float* src = x + ((r * SEQ + t) * CIN + c8);
  const float4 a = *(const float4*)(src);
  const float4 b = *(const float4*)(src + 4);
  v4u o;
  o.x = pack2(a.x, a.y);
  o.y = pack2(a.z, a.w);
  o.z = pack2(b.x, b.y);
  o.w = pack2(b.z, b.w);
  *(v4u*)(xbf + ((t * BATCH + r) * CIN + c8)) = o;
}

// Persistent sLSTM scan for one layer. WG owns hidden units [wg*32, wg*32+32) and
// all 4 gates for them; 4 waves, each holding a 16x16 tile of every gate plus its
// register-resident cell state for all 2048 steps. Layer 1 WGs additionally wait
// on layer 0's per-step counter before consuming h1[t] (layer-pipelined wavefront).
template<int LAYER, int K_IN>
__device__ __forceinline__ void scan_impl(
    const unsigned short* __restrict__ zin,  // [SEQ,B,K_IN] bf16 input sequence
    unsigned short* __restrict__ h1ws,       // [SEQ,B,HID] bf16 (written by L0)
    const float* __restrict__ Wg,            // [K_IN, 4H]
    const float* __restrict__ Rg,            // [HID, 4H]
    const float* __restrict__ bg,            // [4H]
    unsigned short* __restrict__ hbuf,       // [2][B][HID] bf16 double buffer
    unsigned int* cntOwn,                    // [SEQ] this layer's barrier counters
    unsigned int* cntDep,                    // [SEQ] producer layer's counters (L1)
    float* __restrict__ out,                 // L1: [B,SEQ,HID] fp32
    int wg, char* smem)
{
  constexpr int NKI = K_IN / 32;      // input-part K chunks
  constexpr int NKH = HID / 32;       // recurrent-part K chunks
  constexpr int NK  = NKI + NKH;
  constexpr int NBLK = NK / 4;        // pipeline blocks of 4 chunks (6 or 8)
  constexpr int NFRAG = NK * 8;
  unsigned short* lw = (unsigned short*)smem;

  const int tid  = threadIdx.x;
  const int lane = tid & 31;
  const int wave = tid >> 5;
  const int mrow = (wave & 1) * 16;   // batch-row tile base
  const int hsub = wave >> 1;         // which 16-wide hidden subtile
  const int wgbase = wg * HB;
  const int n  = lane & 15;
  const int hi = lane >> 4;

  // One-time: stage W/R slice into LDS, pre-swizzled into WMMA B-fragment layout.
// Fragment f = kc*8 + gate*2 + hsf : 32 lanes x 16 bf16 (lane = N col, K = e<8 ? hi*8+e : 16+hi*8+e-8).
  for (int f = wave; f < NFRAG; f += 4) {
    const int kc = f >> 3, ct = f & 7, gate = ct >> 1, hsf = ct & 1;
    const int col = gate * HID + wgbase + hsf * 16 + n;
    const int kb  = kc * 32 + hi * 8;
    unsigned short* dst = lw + (size_t)f * 512 + lane * 16;
#pragma unroll
    for (int e = 0; e < 16; ++e) {
      const int k = kb + (e < 8 ? e : e + 8);
      const float v = (k < K_IN) ? Wg[(size_t)k * FH + col]
                                 : Rg[(size_t)(k - K_IN) * FH + col];
      dst[e] = f2bf(v);
    }
  }
  __syncthreads();

  float bias[4];
#pragma unroll
  for (int g = 0; g < 4; ++g) bias[g] = bg[g * HID + wgbase + hsub * 16 + n];

  v8f zero8;                          // loop-invariant zero C operand (no per-step
#pragma unroll
  for (int j = 0; j < 8; ++j) zero8[j] = 0.f;   // accumulator init chains)

  v8f cst;                            // cell state tile, register-resident all scan
#pragma unroll
  for (int j = 0; j < 8; ++j) cst[j] = 0.f;

  const int r    = mrow + n;                  // batch row this lane supplies to A
  const int colh = wgbase + hsub * 16 + n;    // hidden column this lane writes

  for (int t = 0; t < SEQ; ++t) {
    // L1: wait for the producer layer to publish h1[t] (release by cnt0 arrive).
    if (LAYER == 1) {
      if (tid == 0) {
        volatile unsigned* vd = cntDep + t;
        while (*vd < (unsigned)NWG) __builtin_amdgcn_s_sleep(1);
      }
      __syncthreads();
      __threadfence();   // acquire: invalidate near caches before reading h1[t]
    }

    const unsigned short* hprev = hbuf + ((t & 1) ? BH : 0);
    unsigned short* hnext = hbuf + ((t & 1) ? 0 : BH);

    // Per-lane A sources (ISA 7.12.2 16-bit A layout: the lane's 16 K-elements
    // are two contiguous 8-element runs at +8*hi and +16+8*hi).
    const unsigned short* srcI = zin + ((size_t)t * BATCH + r) * K_IN + 8 * hi;
    const unsigned short* srcH = hprev + (size_t)r * HID + 8 * hi;
    if (LAYER == 0 && t + 1 < SEQ)
      __builtin_prefetch(srcI + (size_t)BATCH * K_IN, 0, 3);

    // Software-pipelined fused-K loop: blocks of 4 chunks, A double-buffered;
    // sched_barrier pins block i+1's global loads ahead of block i's 16 WMMAs so
    // the load latency is covered (1 wave/SIMD has no other latency hiding).
    v8f acc[4];
    AFrag abuf[2][4];
#pragma unroll
    for (int q = 0; q < 4; ++q) {
      const unsigned short* s = (q < NKI) ? srcI + q * 32 : srcH + (q - NKI) * 32;
      abuf[0][q].q[0] = *(const v4u*)(s);
      abuf[0][q].q[1] = *(const v4u*)(s + 16);
    }
#pragma unroll
    for (int blk = 0; blk < NBLK; ++blk) {
      const int cur = blk & 1;
      const int nxt = cur ^ 1;
      if (blk + 1 < NBLK) {
#pragma unroll
        for (int q = 0; q < 4; ++q) {
          const int kc = (blk + 1) * 4 + q;
          const unsigned short* s =
              (kc < NKI) ? srcI + kc * 32 : srcH + (kc - NKI) * 32;
          abuf[nxt][q].q[0] = *(const v4u*)(s);
          abuf[nxt][q].q[1] = *(const v4u*)(s + 16);
        }
      }
      __builtin_amdgcn_sched_barrier(0);  // keep prefetch loads above the WMMAs
#pragma unroll
      for (int q = 0; q < 4; ++q) {
        const int kc = blk * 4 + q;
#pragma unroll
        for (int g = 0; g < 4; ++g) {
          BFrag bf;
          const v4u* bq = (const v4u*)(smem +
              (size_t)(kc * 8 + g * 2 + hsub) * 1024 + lane * 32);
          bf.q[0] = bq[0];
          bf.q[1] = bq[1];
          // First chunk seeds the chain from the invariant zero vector.
          acc[g] = __builtin_amdgcn_wmma_f32_16x16x32_bf16(
              false, abuf[cur][q].v, false, bf.v, (short)0,
              (kc == 0) ? zero8 : acc[g], false, false);
        }
      }
    }

    // Gate nonlinearities + state update, fully in registers (D-layout
    // elementwise); bias folded here instead of accumulator initialization.
#pragma unroll
    for (int j = 0; j < 8; ++j) {
      const float iv = acc[0][j] + bias[0];
      const float fv = acc[1][j] + bias[1];
      const float zv = acc[2][j] + bias[2];
      const float ov = acc[3][j] + bias[3];
      const float ms = fmaxf(iv, fv);
      const float ie = __expf(iv - ms);
      const float fe = __expf(fv - ms);
      const float ze = fast_tanh(zv);
      const float oe = fast_sig(ov);
      const float c2 = fe * cst[j] + ie * ze;
      cst[j] = c2;
      const float hv = oe * fast_tanh(c2);
      const int row = mrow + j + hi * 8;        // D layout: lane>=16 -> M+8
      const unsigned short hb = f2bf(hv);
      hnext[(size_t)row * HID + colh] = hb;
      if (LAYER == 0)
        h1ws[((size_t)t * BATCH + row) * HID + colh] = hb;
      else
        out[((size_t)row * SEQ + t) * HID + colh] = hv;
    }

    // Release + step barrier among this layer's 16 WGs. For L0, the same counter
    // is the "h1[t] ready" signal that unblocks layer 1's step t.
    __threadfence();
    __syncthreads();
    if (tid == 0) {
      atomicAdd(&cntOwn[t], 1u);
      volatile unsigned* vc = cntOwn + t;
      while (*vc < (unsigned)NWG) __builtin_amdgcn_s_sleep(1);
    }
    __syncthreads();
    __threadfence();
  }
}

// One fused persistent kernel: blocks [0,16) run layer 0, blocks [16,32) run
// layer 1, pipelined one timestep apart -> ~2049 sequential steps instead of 4096.
__global__ __launch_bounds__(128, 1)
void slstm_fused(const unsigned short* __restrict__ xbf,
                 unsigned short* __restrict__ h1ws,
                 const float* __restrict__ W0, const float* __restrict__ R0,
                 const float* __restrict__ b0,
                 const float* __restrict__ W1, const float* __restrict__ R1,
                 const float* __restrict__ b1,
                 unsigned short* __restrict__ hbuf,   // [2 layers][2][B][HID]
                 unsigned int* cnt,                   // [2][SEQ]
                 float* __restrict__ out)
{
  extern __shared__ char smem[];
  if (blockIdx.x < (unsigned)NWG) {
    scan_impl<0, CIN>(xbf, h1ws, W0, R0, b0,
                      hbuf, cnt, nullptr, nullptr, blockIdx.x, smem);
  } else {
    scan_impl<1, HID>(h1ws, nullptr, W1, R1, b1,
                      hbuf + 2 * BH, cnt + SEQ, cnt, out, blockIdx.x - NWG, smem);
  }
}

extern "C" void kernel_launch(void* const* d_in, const int* in_sizes, int n_in,
                              void* d_out, int out_size, void* d_ws, size_t ws_size,
                              hipStream_t stream) {
  (void)in_sizes; (void)n_in; (void)out_size; (void)ws_size;
  const float* x  = (const float*)d_in[0];
  const float* W0 = (const float*)d_in[1];
  const float* R0 = (const float*)d_in[2];
  const float* b0 = (const float*)d_in[3];
  const float* W1 = (const float*)d_in[4];
  const float* R1 = (const float*)d_in[5];
  const float* b1 = (const float*)d_in[6];
  float* out = (float*)d_out;

  // ws: [xbf SEQ*B*CIN bf16] | [h1 SEQ*B*HID bf16] | [4*BH bf16 h bufs] | [2*SEQ u32]
  unsigned short* xbf  = (unsigned short*)d_ws;
  unsigned short* h1ws = xbf + (size_t)SEQ * BATCH * CIN;
  unsigned short* hbuf = h1ws + (size_t)SEQ * BATCH * HID;
  unsigned int*   cnt  = (unsigned int*)(hbuf + (size_t)4 * BH);

  hipLaunchKernelGGL(init_ws, dim3(64), dim3(256), 0, stream, hbuf, cnt);
  {
    const size_t total = (size_t)BATCH * SEQ * (CIN / 8);
    const int blk = 256;
    const int grd = (int)((total + blk - 1) / blk);
    hipLaunchKernelGGL(prep_x, dim3(grd), dim3(blk), 0, stream, x, xbf);
  }

  // LDS = max of the two layers' pre-swizzled weight slices (L1: 256KB).
  constexpr int NK1 = (HID + HID) / 32;
  const size_t lds = (size_t)NK1 * 8 * 1024;
  (void)hipFuncSetAttribute((const void*)slstm_fused,
                            hipFuncAttributeMaxDynamicSharedMemorySize, (int)lds);

  slstm_fused<<<2 * NWG, 128, lds, stream>>>(xbf, h1ws, W0, R0, b0, W1, R1, b1,
                                             hbuf, cnt, out);
}
